// PhysicsInformedEvolution_16303695855858
// MI455X (gfx1250) — compile-verified, matrix operand order checked by reading, not run on hardware
//
#include <hip/hip_runtime.h>
#include <hip/hip_fp16.h>
#include <math.h>

typedef __attribute__((ext_vector_type(16))) _Float16 v16h;
typedef __attribute__((ext_vector_type(8)))  _Float16 v8h;
typedef __attribute__((ext_vector_type(8)))  float    v8f;

// Problem constants (from reference)
#define B_TOT   4096
#define ROWS    64       // batch rows per workgroup (4 M-tiles)
#define NBLK    (B_TOT / ROWS)
#define TSTEPS  128
#define LAT     64
#define IC_OFF  50.0f
#define LOG2000 7.6009024595420822f

// ---------------------------------------------------------------------------
// Weight pack kernel: build WMMA B-fragments (K x N = W^T tiles) in the
// per-lane layout of V_WMMA_F32_16X16X32_F16 (ISA 7.12.2, 16-bit operand).
// Fragment (nt,kt) stored linearly: dst[((nt*KT+kt)*32 + lane)*16 + e].
// Element (lane,e) -> N = nt*16 + (lane&15),
//                     K = kt*32 + (e&7) + 16*(e>>3) + 8*(lane>>4).
// Virtual K axis: [0,Ka) from srcA (row-major O x Ka); [Kslot,Kslot+Kb) from
// srcB (row-major O x Kb); zero elsewhere (padding).
// ---------------------------------------------------------------------------
__global__ void pack_wmma_b(_Float16* dst,
                            const float* srcA, int Ka,
                            int Kslot, const float* srcB, int Kb,
                            int O, int KT, int total)
{
    int tid = blockIdx.x * blockDim.x + threadIdx.x;
    if (tid >= total) return;
    int e    = tid & 15;
    int lane = (tid >> 4) & 31;
    int frag = tid >> 9;
    int kt   = frag % KT;
    int nt   = frag / KT;
    int N = nt * 16 + (lane & 15);
    int K = kt * 32 + (e & 7) + 16 * (e >> 3) + ((lane >> 4) << 3);
    float v = 0.f;
    if (N < O) {
        if (K < Ka)                                   v = srcA[N * Ka + K];
        else if (srcB && K >= Kslot && K < Kslot+Kb)  v = srcB[N * Kb + (K - Kslot)];
    }
    dst[tid] = (_Float16)v;
}

__global__ void bias_combine(float* dst, const float* a, const float* b, int n)
{
    int i = blockIdx.x * blockDim.x + threadIdx.x;
    if (i < n) dst[i] = a[i] + b[i];
}

// ---------------------------------------------------------------------------
// Device helpers
// ---------------------------------------------------------------------------
__device__ __forceinline__ v16h load_a_frag(const _Float16* base, int ld,
                                            int row0, int k0, int lane)
{
    // 16-bit A 16x32 layout: M=lane&15; halves e=0..7 -> K=k0+off+e,
    // e=8..15 -> K=k0+off+16+(e-8), off = 8*(lane>>4).
    const _Float16* p = base + (row0 + (lane & 15)) * ld + k0 + ((lane >> 4) << 3);
    v8h lo = *(const v8h*)(p);
    v8h hi = *(const v8h*)(p + 16);
    return __builtin_shufflevector(lo, hi, 0,1,2,3,4,5,6,7,8,9,10,11,12,13,14,15);
}

__device__ __forceinline__ v16h load_b_frag(const _Float16* wp, int frag, int lane)
{
    return *(const v16h*)(wp + (size_t)frag * 512 + lane * 16);
}

__device__ __forceinline__ float sigf(float x) { return 1.f / (1.f + __expf(-x)); }

// ---------------------------------------------------------------------------
// One LSTM layer, column-split: this wave owns hidden columns
// [jt*16, jt*16+16) (with their i/f/g/o gate tiles) for ALL 64 rows.
// A = [A0|A1] (virtual K=256, each 64x128 f16 LDS), W packed NT=32,KT=8.
// cst = 4 c-state tiles (v8f, accumulator layout). Hdst: 64x128 f16 LDS.
// B fragments are loaded once per (gate,kt) and reused across all 4 M-tiles;
// waves load disjoint fragments -> no duplicated weight traffic inside a WG.
// ---------------------------------------------------------------------------
__device__ __forceinline__ void lstm_layer(const _Float16* A0, const _Float16* A1,
                                           const _Float16* Wp, const float* bias,
                                           v8f* cst, _Float16* Hdst,
                                           int lane, int jt)
{
    const int n = lane & 15;
    v8f acc[4][4];   // [gate][mtile]
    #pragma unroll
    for (int g = 0; g < 4; ++g) {
        float bv = bias[g * 128 + jt * 16 + n];
        #pragma unroll
        for (int m = 0; m < 4; ++m)
            #pragma unroll
            for (int e = 0; e < 8; ++e) acc[g][m][e] = bv;
    }

    #pragma unroll
    for (int kt = 0; kt < 8; ++kt) {
        const _Float16* Ab = (kt < 4) ? A0 : A1;
        const int k0 = (kt & 3) * 32;
        v16h Bg[4];
        #pragma unroll
        for (int g = 0; g < 4; ++g)
            Bg[g] = load_b_frag(Wp, (g * 8 + jt) * 8 + kt, lane);
        #pragma unroll
        for (int m = 0; m < 4; ++m) {
            v16h a = load_a_frag(Ab, 128, m * 16, k0, lane);
            #pragma unroll
            for (int g = 0; g < 4; ++g)
                acc[g][m] = __builtin_amdgcn_wmma_f32_16x16x32_f16(
                    false, a, false, Bg[g], (short)0, acc[g][m], false, false);
        }
    }

    #pragma unroll
    for (int m = 0; m < 4; ++m) {
        v8f c = cst[m];
        #pragma unroll
        for (int e = 0; e < 8; ++e) {
            float cn = sigf(acc[1][m][e]) * c[e] + sigf(acc[0][m][e]) * tanhf(acc[2][m][e]);
            c[e] = cn;
            float h = sigf(acc[3][m][e]) * tanhf(cn);
            int row = m * 16 + e + ((lane >> 4) << 3);
            Hdst[row * 128 + jt * 16 + n] = (_Float16)h;
        }
        cst[m] = c;
    }
}

// Small GEMM + ReLU, column-split: this wave computes output tile nt for all
// 4 M-tiles. Dst = relu(A @ Wp^T + bias).
template<int KT, int LDA, int LDD>
__device__ __forceinline__ void mlp_tiles(const _Float16* A, const _Float16* Wp,
                                          const float* bias, _Float16* Dst,
                                          int nt, int lane)
{
    const int n = lane & 15;
    float bv = bias[nt * 16 + n];
    v8f acc[4];
    #pragma unroll
    for (int m = 0; m < 4; ++m)
        #pragma unroll
        for (int e = 0; e < 8; ++e) acc[m][e] = bv;

    #pragma unroll
    for (int kt = 0; kt < KT; ++kt) {
        v16h b = load_b_frag(Wp, nt * KT + kt, lane);
        #pragma unroll
        for (int m = 0; m < 4; ++m) {
            v16h a = load_a_frag(A, LDA, m * 16, kt * 32, lane);
            acc[m] = __builtin_amdgcn_wmma_f32_16x16x32_f16(
                false, a, false, b, (short)0, acc[m], false, false);
        }
    }
    #pragma unroll
    for (int m = 0; m < 4; ++m)
        #pragma unroll
        for (int e = 0; e < 8; ++e) {
            int row = m * 16 + e + ((lane >> 4) << 3);
            Dst[row * LDD + nt * 16 + n] = (_Float16)fmaxf(acc[m][e], 0.f);
        }
}

// ---------------------------------------------------------------------------
// Persistent main kernel: 64 blocks x 256 threads (8 wave32 waves); each block
// owns 64 batch rows and iterates all 128 timesteps. Wave w owns hidden
// columns [16w,16w+16): its c-states live in VGPRs in WMMA accumulator layout
// and its weight fragments are disjoint from other waves'. h-states are
// double-buffered f16 in LDS.
// ---------------------------------------------------------------------------
__global__ void __launch_bounds__(256) pie_main(
    const float* __restrict__ init_states, const float* __restrict__ latent,
    const _Float16* __restrict__ W0p_, const _Float16* __restrict__ W1p_,
    const _Float16* __restrict__ Wg1p_, const _Float16* __restrict__ Wp1p_,
    const float* __restrict__ b0, const float* __restrict__ b1,
    const float* __restrict__ bg1, const float* __restrict__ bp1,
    const float* __restrict__ wg2, const float* __restrict__ bg2,
    const float* __restrict__ wp2, const float* __restrict__ bp2,
    float* __restrict__ out)
{
    extern __shared__ char smem[];
    _Float16* sh_X  = (_Float16*)smem;            // 64 x 128  (x padded to 128)
    _Float16* sh_H0 = sh_X  + ROWS * 128;         // 2 x (64 x 128) double buffer
    _Float16* sh_H1 = sh_H0 + 2 * ROWS * 128;     // 2 x (64 x 128)
    _Float16* sh_R  = sh_H1 + 2 * ROWS * 128;     // 64 x 96   (rate input padded)
    _Float16* sh_T  = sh_R  + ROWS * 96;          // 64 x 128  scratch
    float*  sh_cur  = (float*)(sh_T + ROWS * 128);
    float*  sh_rate = sh_cur + ROWS;
    float*  sh_freq = sh_rate + ROWS;             // 16

    const int tid   = threadIdx.x;
    const int lane  = tid & 31;
    const int wave  = tid >> 5;                   // == jt / nt column tile
    const int rbase = blockIdx.x * ROWS;

    // Local copies of weight pointers; made opaque each timestep so the
    // compiler cannot LICM-hoist 100s of fragment loads into scratch spills.
    const _Float16* W0p  = W0p_;
    const _Float16* W1p  = W1p_;
    const _Float16* Wg1p = Wg1p_;
    const _Float16* Wp1p = Wp1p_;

    // ---- one-time init: x = [cur | latent | emb | 0], R = [latent | cur | 0]
    for (int e = tid; e < ROWS * 128; e += 256) {
        int row = e >> 7, col = e & 127;
        float v = (col >= 1 && col < 65) ? latent[(rbase + row) * LAT + (col - 1)] : 0.f;
        sh_X[e] = (_Float16)v;
    }
    for (int e = tid; e < ROWS * 96; e += 256) {
        int row = e / 96, col = e - row * 96;
        float v = (col < 64) ? latent[(rbase + row) * LAT + col] : 0.f;
        sh_R[e] = (_Float16)v;
    }
    for (int e = tid; e < 2 * ROWS * 128; e += 256) {
        sh_H0[e] = (_Float16)0.f;
        sh_H1[e] = (_Float16)0.f;
    }
    if (tid < ROWS) sh_cur[tid] = init_states[(rbase + tid) * 50 + 49];
    if (tid < 16)   sh_freq[tid] = __expf(-LOG2000 * (float)tid * (1.f / 16.f));

    v8f c0[4], c1[4];
    #pragma unroll
    for (int m = 0; m < 4; ++m)
        #pragma unroll
        for (int e = 0; e < 8; ++e) { c0[m][e] = 0.f; c1[m][e] = 0.f; }
    __syncthreads();

    int buf = 0;
    #pragma unroll 1
    for (int t = 0; t < TSTEPS; ++t) {
        // keep weight pointers loop-variant (defeat LICM -> scratch spills)
        asm volatile("" : "+s"(W0p), "+s"(W1p), "+s"(Wg1p), "+s"(Wp1p));

        // (a) refresh cur column + sinusoidal time embedding (x cols 65..96)
        if (tid < ROWS) {
            _Float16 hc = (_Float16)sh_cur[tid];
            sh_X[tid * 128 + 0] = hc;
            sh_R[tid * 96 + 64] = hc;
        }
        {
            float cyc = IC_OFF + (float)t;
            for (int e = tid; e < ROWS * 32; e += 256) {
                int row = e >> 5, k = e & 31;
                float arg = cyc * sh_freq[k & 15];
                float v = (k < 16) ? __sinf(arg) : __cosf(arg);
                sh_X[row * 128 + 65 + k] = (_Float16)v;
            }
        }
        __syncthreads();

        // (b) rate head: relu(R @ Wg1^T + bg1) -> sh_T; softplus(.@wg2 + bg2)
        mlp_tiles<3, 96, 128>(sh_R, Wg1p, bg1, sh_T, wave, lane);
        __syncthreads();
        if (tid < ROWS) {
            float s = bg2[0];
            #pragma unroll 4
            for (int j = 0; j < 128; ++j) s += (float)sh_T[tid * 128 + j] * wg2[j];
            sh_rate[tid] = (s > 20.f) ? s : logf(1.f + __expf(s));
        }
        __syncthreads();

        // (c) LSTM layer 0: gates = [x|h0] @ W0^T + b0
        lstm_layer(sh_X, sh_H0 + buf * (ROWS * 128), W0p, b0, c0,
                   sh_H0 + (buf ^ 1) * (ROWS * 128), lane, wave);
        __syncthreads();

        // (d) LSTM layer 1: gates = [h0n|h1] @ W1^T + b1
        lstm_layer(sh_H0 + (buf ^ 1) * (ROWS * 128), sh_H1 + buf * (ROWS * 128),
                   W1p, b1, c1, sh_H1 + (buf ^ 1) * (ROWS * 128), lane, wave);
        __syncthreads();

        // (e) delta head: relu(h1n @ Wp1^T + bp1) -> sh_T (64 x 64)
        if (wave < 4)
            mlp_tiles<4, 128, 64>(sh_H1 + (buf ^ 1) * (ROWS * 128), Wp1p, bp1,
                                  sh_T, wave, lane);
        __syncthreads();

        // (f) finish: delta = relu(.@wp2 + bp2); nxt = cur + delta*rate
        if (tid < ROWS) {
            float d = bp2[0];
            #pragma unroll 4
            for (int j = 0; j < 64; ++j) d += (float)sh_T[tid * 64 + j] * wp2[j];
            d = fmaxf(d, 0.f);
            float nxt = sh_cur[tid] + d * sh_rate[tid];
            sh_cur[tid] = nxt;
            out[(size_t)(rbase + tid) * TSTEPS + t] = nxt;
        }
        __syncthreads();
        buf ^= 1;
    }
}

// ---------------------------------------------------------------------------
// Host launcher
// ---------------------------------------------------------------------------
extern "C" void kernel_launch(void* const* d_in, const int* in_sizes, int n_in,
                              void* d_out, int out_size, void* d_ws, size_t ws_size,
                              hipStream_t stream)
{
    (void)in_sizes; (void)n_in; (void)out_size; (void)ws_size;
    const float* init_states = (const float*)d_in[0];
    const float* latent      = (const float*)d_in[1];
    const float* w_ih0 = (const float*)d_in[3];
    const float* w_hh0 = (const float*)d_in[4];
    const float* b_ih0 = (const float*)d_in[5];
    const float* b_hh0 = (const float*)d_in[6];
    const float* w_ih1 = (const float*)d_in[7];
    const float* w_hh1 = (const float*)d_in[8];
    const float* b_ih1 = (const float*)d_in[9];
    const float* b_hh1 = (const float*)d_in[10];
    const float* wp1   = (const float*)d_in[11];
    const float* bp1   = (const float*)d_in[12];
    const float* wp2   = (const float*)d_in[13];
    const float* bp2   = (const float*)d_in[14];
    const float* wg1   = (const float*)d_in[15];
    const float* bg1   = (const float*)d_in[16];
    const float* wg2   = (const float*)d_in[17];
    const float* bg2   = (const float*)d_in[18];
    float* out = (float*)d_out;

    // Workspace layout (bytes)
    char* ws = (char*)d_ws;
    _Float16* W0p  = (_Float16*)(ws + 0);        // 32*8 frags * 1024B = 262144
    _Float16* W1p  = (_Float16*)(ws + 262144);   // 262144
    _Float16* Wg1p = (_Float16*)(ws + 524288);   // 8*3 frags  = 24576
    _Float16* Wp1p = (_Float16*)(ws + 548864);   // 4*4 frags  = 16384
    float*    b0   = (float*)   (ws + 565248);   // 512 f32
    float*    b1   = (float*)   (ws + 567296);   // 512 f32

    // Pack weights into WMMA B-fragment layout (f16)
    {
        int tot = 32 * 8 * 512;   // W0cat: O=512, K=256 (x[0..96]|pad|h[128..255])
        pack_wmma_b<<<(tot + 255) / 256, 256, 0, stream>>>(W0p, w_ih0, 97, 128, w_hh0, 128, 512, 8, tot);
        pack_wmma_b<<<(tot + 255) / 256, 256, 0, stream>>>(W1p, w_ih1, 128, 128, w_hh1, 128, 512, 8, tot);
        int totg = 8 * 3 * 512;   // Wg1: O=128, K=65 -> pad 96
        pack_wmma_b<<<(totg + 255) / 256, 256, 0, stream>>>(Wg1p, wg1, 65, 0, nullptr, 0, 128, 3, totg);
        int totp = 4 * 4 * 512;   // Wp1: O=64, K=128
        pack_wmma_b<<<(totp + 255) / 256, 256, 0, stream>>>(Wp1p, wp1, 128, 0, nullptr, 0, 64, 4, totp);
        bias_combine<<<2, 256, 0, stream>>>(b0, b_ih0, b_hh0, 512);
        bias_combine<<<2, 256, 0, stream>>>(b1, b_ih1, b_hh1, 512);
    }

    // LDS: X(16K) + H0(32K) + H1(32K) + R(12K) + T(16K) + cur/rate/freq
    size_t lds = (size_t)(ROWS*128 + 2*ROWS*128 + 2*ROWS*128 + ROWS*96 + ROWS*128) * 2
               + (2 * ROWS + 16) * sizeof(float);

    pie_main<<<NBLK, 256, lds, stream>>>(init_states, latent,
                                         W0p, W1p, Wg1p, Wp1p,
                                         b0, b1, bg1, bp1,
                                         wg2, bg2, wp2, bp2, out);
}
